// DetrTransformerDecoderLayer_41180146434516
// MI455X (gfx1250) — compile-verified, matrix operand order checked by reading, not run on hardware
//
#include <hip/hip_runtime.h>
#include <hip/hip_bf16.h>

// ---------------- problem constants (match reference) ----------------
#define NQ   900
#define BSZ  16
#define EMB  256
#define NHD  8
#define HD   32
#define NP   4
#define DFF  512
#define NVAL 10000
#define NQB  (NQ * BSZ)       // 14400 rows in (q, b) flattening
#define LNEPS 1e-5f

typedef __attribute__((ext_vector_type(16))) _Float16 v16h;
typedef __attribute__((ext_vector_type(8)))  float    v8f;
typedef __attribute__((ext_vector_type(4)))  unsigned uint32x4;
typedef __attribute__((ext_vector_type(8)))  int      int32x8;
typedef __attribute__((ext_vector_type(4)))  int      int32x4;

#if __has_builtin(__builtin_amdgcn_tensor_load_to_lds) && \
    __has_builtin(__builtin_amdgcn_s_wait_tensorcnt)
#define USE_TDM 1
#else
#define USE_TDM 0
#endif

union U16x16 { v16h v; unsigned u[8]; };

__device__ inline unsigned pack2(float a, float b) {
  union { _Float16 h[2]; unsigned u; } p;
  p.h[0] = (_Float16)a; p.h[1] = (_Float16)b;
  return p.u;
}

// generic shared pointer -> LDS byte offset (flat LDS aperture: addr[31:0])
__device__ inline unsigned lds_off(const void* p) {
  return (unsigned)(size_t)p;
}

// Load a 16x32 f16 A/B WMMA fragment from a row-major f16 LDS tile.
// Layout per CDNA5 ISA 7.12.2 (16-bit A-matrix 16x32):
//   lanes 0-15 -> rows 0-15, K chunks 0-7 / 16-23; lanes 16-31 -> K+8 chunks.
__device__ inline v16h ld_frag(const _Float16* base, int stride_h, int lane) {
  const int r  = lane & 15;
  const int hi = (lane >> 4) & 1;
  U16x16 f;
  const unsigned char* row = (const unsigned char*)(base + r * stride_h);
#pragma unroll
  for (int v = 0; v < 8; ++v) {
    const int kbase = ((v >> 2) << 4) + (hi << 3) + ((v & 3) << 1); // even K
    f.u[v] = *(const unsigned*)(row + kbase * 2);
  }
  return f.v;
}

__device__ inline v8f wmma16(v16h a, v16h b, v8f c) {
  return __builtin_amdgcn_wmma_f32_16x16x32_f16(false, a, false, b,
                                                (short)0, c, false, false);
}

#if USE_TDM
// TDM 2D tile load: tile_rows x 32 dwords from a row-major fp32 matrix with
// row stride `stride_e` elements, into LDS at lds_byte with +2-dword padding
// after every 32 dwords (=> stride-34 fp32 rows, conflict-free gathers).
// Rows/cols beyond rows_avail/cols_avail are zero-filled by TDM OOB handling.
// Descriptor bitfields per CDNA5 ISA sec. 8.3/8.4; 6-arg builtin (clang-23
// form): groups 2/3 + extra vector unused for 2D tiles.
__device__ inline void tdm_load_2d(unsigned lds_byte, const void* gptr,
                                   int rows_avail, int cols_avail,
                                   int stride_e, int tile_rows) {
  unsigned long long ga = (unsigned long long)(size_t)gptr;
  uint32x4 g0;
  g0[0] = 1u;                                        // count=1, user mode
  g0[1] = lds_byte;                                  // lds_addr
  g0[2] = (unsigned)(ga & 0xFFFFFFFFu);              // global_addr[31:0]
  g0[3] = (unsigned)((ga >> 32) & 0x1FFFFFFu)        // global_addr[56:32]
        | (2u << 30);                                // type=2 (image)
  const unsigned d0 = (unsigned)cols_avail;          // tensor_dim0
  const unsigned d1 = (unsigned)rows_avail;          // tensor_dim1
  int32x8 g1;
  g1[0] = (int)((2u << 16)    // data_size = 4 bytes
              | (1u << 20)    // pad_enable
              | (4u << 22)    // pad_interval: every 32 dwords
              | (1u << 25));  // pad_amount: 2 dwords
  g1[1] = (int)((d0 & 0xFFFFu) << 16);               // dim0[15:0] @ bits 63:48
  g1[2] = (int)((d0 >> 16) | ((d1 & 0xFFFFu) << 16));// dim0[31:16], dim1[15:0]
  g1[3] = (int)((d1 >> 16) | (32u << 16));           // dim1[31:16], tile_dim0=32
  g1[4] = (int)(unsigned)tile_rows;                  // tile_dim1 (tile_dim2=0)
  g1[5] = (int)(unsigned)stride_e;                   // dim0_stride[31:0]
  g1[6] = 0;                                         // dim0_stride[47:32]
  g1[7] = 0;
  const int32x4 z4 = {0, 0, 0, 0};
  const int32x8 z8 = {0, 0, 0, 0, 0, 0, 0, 0};
  __builtin_amdgcn_tensor_load_to_lds(g0, g1, z4, z4, z8, 0);
}
#endif

// ---------------- generic WMMA GEMM: C[M,N] = A[M,K] @ W[N,K]^T + bias ----
#define BM 128
#define BN 64
#define BK 32
#define SF 34      // fp32 LDS row stride (32 + 2 pad): 8B aligned, no conflicts
#define AST 34     // f16 LDS row stride in halves
#define AST_U 17

__global__ __launch_bounds__(256)
void gemm_f16_kernel(const float* __restrict__ A, const float* __restrict__ W,
                     const float* __restrict__ bias, void* __restrict__ Cout,
                     int M, int N, int K, int relu, int store_half)
{
  __shared__ float    As32[BM * SF];     // 17408 B (TDM destination)
  __shared__ float    Bs32[BN * SF];     //  8704 B
  __shared__ unsigned As16[BM * AST_U];  //  8704 B (packed f16 tiles)
  __shared__ unsigned Bs16[BN * AST_U];  //  4352 B

  const int tid  = threadIdx.x;
  const int lane = tid & 31;
  const int wave = tid >> 5;
  const int wm   = wave & 3;      // 4 M-strips of 32 rows
  const int wn   = wave >> 2;     // 2 N-strips of 32 cols
  const int m0   = blockIdx.x * BM;
  const int n0   = blockIdx.y * BN;

  v8f c[2][2] = {};

  for (int k0 = 0; k0 < K; k0 += BK) {
    __syncthreads();              // prev iter's f32/f16 tile reads drained
#if USE_TDM
    // Tensor Data Mover: wave0 DMAs the A tile, wave1 the B tile.
    if (wave == 0) {
      tdm_load_2d(lds_off(As32), A + (size_t)m0 * K + k0, M - m0, K - k0, K, BM);
      __builtin_amdgcn_s_wait_tensorcnt(0);
    } else if (wave == 1) {
      tdm_load_2d(lds_off(Bs32), W + (size_t)n0 * K + k0, N - n0, K - k0, K, BN);
      __builtin_amdgcn_s_wait_tensorcnt(0);
    }
#else
    // de-branched staging: clamped addresses, unconditional (batchable) loads
#pragma unroll
    for (int i = 0; i < 8; ++i) {
      int idx = tid + i * 256;
      int row = idx >> 4;
      int kp  = idx & 15;
      int m   = m0 + row;
      int mc  = m < M ? m : M - 1;
      float2 xy = *(const float2*)(A + (size_t)mc * K + k0 + kp * 2);
      if (m >= M) { xy.x = 0.f; xy.y = 0.f; }
      *(float2*)(&As32[row * SF + kp * 2]) = xy;
    }
#pragma unroll
    for (int i = 0; i < 4; ++i) {
      int idx = tid + i * 256;
      int row = idx >> 4;
      int kp  = idx & 15;
      int n   = n0 + row;
      int nc  = n < N ? n : N - 1;
      float2 xy = *(const float2*)(W + (size_t)nc * K + k0 + kp * 2);
      if (n >= N) { xy.x = 0.f; xy.y = 0.f; }
      *(float2*)(&Bs32[row * SF + kp * 2]) = xy;
    }
#endif
    __syncthreads();              // f32 tiles ready

    // cooperative convert: one v_cvt_pk per element pair, done exactly once
#pragma unroll
    for (int i = 0; i < 8; ++i) {
      int idx = tid + i * 256;
      int row = idx >> 4;
      int kp  = idx & 15;
      float2 xy = *(const float2*)(&As32[row * SF + kp * 2]);
      As16[row * AST_U + kp] = pack2(xy.x, xy.y);
    }
#pragma unroll
    for (int i = 0; i < 4; ++i) {
      int idx = tid + i * 256;
      int row = idx >> 4;
      int kp  = idx & 15;
      float2 xy = *(const float2*)(&Bs32[row * SF + kp * 2]);
      Bs16[row * AST_U + kp] = pack2(xy.x, xy.y);
    }
    __syncthreads();              // f16 tiles ready

    v16h a0 = ld_frag((const _Float16*)As16 + (wm * 32 +  0) * AST, AST, lane);
    v16h a1 = ld_frag((const _Float16*)As16 + (wm * 32 + 16) * AST, AST, lane);
    v16h b0 = ld_frag((const _Float16*)Bs16 + (wn * 32 +  0) * AST, AST, lane);
    v16h b1 = ld_frag((const _Float16*)Bs16 + (wn * 32 + 16) * AST, AST, lane);
    c[0][0] = wmma16(a0, b0, c[0][0]);
    c[0][1] = wmma16(a0, b1, c[0][1]);
    c[1][0] = wmma16(a1, b0, c[1][0]);
    c[1][1] = wmma16(a1, b1, c[1][1]);
  }

  const int hi = (lane >> 4) & 1;
#pragma unroll
  for (int i = 0; i < 2; ++i)
#pragma unroll
    for (int j = 0; j < 2; ++j) {
      int n = n0 + wn * 32 + j * 16 + (lane & 15);
      float bv = (n < N) ? bias[n] : 0.f;
#pragma unroll
      for (int r = 0; r < 8; ++r) {
        int m = m0 + wm * 32 + i * 16 + r + 8 * hi;
        if (m < M && n < N) {
          float v = c[i][j][r] + bv;
          if (relu) v = fmaxf(v, 0.f);
          if (store_half) ((_Float16*)Cout)[(size_t)m * N + n] = (_Float16)v;
          else            ((float*)Cout)[(size_t)m * N + n] = v;
        }
      }
    }
}

// ---------------- fused flash self-attention ----------------
// grid.x = bs*NH (128), grid.y = ceil(NQ/64), block = 128 (4 waves).
#define QBLK 64
#define KBLK 64
#define VST  66    // 64 keys + 2 pad halves

__global__ __launch_bounds__(128)
void flash_attn_kernel(const float* __restrict__ Qp, const float* __restrict__ Kp,
                       const float* __restrict__ Vp, float* __restrict__ Op)
{
  __shared__ unsigned Qs_u[QBLK * AST_U];
  __shared__ unsigned Ks_u[KBLK * AST_U];
  __shared__ _Float16 Vt[HD * VST];          // V transposed: [dim][key]
  __shared__ _Float16 Ps[4 * 16 * VST];      // per-wave P strip

  const int tid  = threadIdx.x;
  const int lane = tid & 31;
  const int wave = tid >> 5;
  const int bh   = blockIdx.x;
  const int b    = bh >> 3;
  const int h    = bh & 7;
  const int qb0  = blockIdx.y * QBLK;
  const float scale = 0.17677669529663687f;  // 1/sqrt(HD)

  // stage Q block (pre-scaled) once; clamped unconditional loads
#pragma unroll
  for (int i = 0; i < 8; ++i) {
    int idx = tid + i * 128;
    int row = idx >> 4;
    int kp  = idx & 15;
    int q   = qb0 + row;
    int qc  = q < NQ ? q : NQ - 1;
    float2 xy = *(const float2*)(Qp + ((size_t)qc * BSZ + b) * EMB + h * HD + kp * 2);
    bool ok = q < NQ;
    Qs_u[row * AST_U + kp] = pack2(ok ? xy.x * scale : 0.f,
                                   ok ? xy.y * scale : 0.f);
  }
  __syncthreads();
  v16h qf = ld_frag((const _Float16*)Qs_u + wave * 16 * AST, AST, lane);

  v8f o0 = {}, o1 = {};
  float m_s[8], l_s[8];
#pragma unroll
  for (int r = 0; r < 8; ++r) { m_s[r] = -1e30f; l_s[r] = 0.f; }

  const int hi = (lane >> 4) & 1;
  _Float16* Pw = Ps + wave * 16 * VST;

  const int nkb = (NQ + KBLK - 1) / KBLK;
  for (int kb = 0; kb < nkb; ++kb) {
    __syncthreads();
    // stage K block [64 keys][32 dims]
#pragma unroll
    for (int i = 0; i < 8; ++i) {
      int idx = tid + i * 128;
      int row = idx >> 4;
      int kp  = idx & 15;
      int kq  = kb * KBLK + row;
      int kqc = kq < NQ ? kq : NQ - 1;
      float2 xy = *(const float2*)(Kp + ((size_t)kqc * BSZ + b) * EMB + h * HD + kp * 2);
      bool ok = kq < NQ;
      Ks_u[row * AST_U + kp] = pack2(ok ? xy.x : 0.f, ok ? xy.y : 0.f);
    }
    // stage V transposed [32 dims][64 keys]
#pragma unroll
    for (int i = 0; i < 16; ++i) {
      int idx = tid + i * 128;
      int d   = idx & 31;
      int kl  = idx >> 5;
      int kq  = kb * KBLK + kl;
      int kqc = kq < NQ ? kq : NQ - 1;
      float v = Vp[((size_t)kqc * BSZ + b) * EMB + h * HD + d];
      Vt[d * VST + kl] = (_Float16)(kq < NQ ? v : 0.f);
    }
    __syncthreads();

    // S strip = Q(16x32) @ K^T(32x64) : 4 WMMAs
    v8f s[4];
#pragma unroll
    for (int t = 0; t < 4; ++t) {
      v16h kfr = ld_frag((const _Float16*)Ks_u + t * 16 * AST, AST, lane);
      v8f z = {};
      s[t] = wmma16(qf, kfr, z);
    }
    // mask padded keys
#pragma unroll
    for (int t = 0; t < 4; ++t) {
      int kq = kb * KBLK + t * 16 + (lane & 15);
      if (kq >= NQ) {
#pragma unroll
        for (int r = 0; r < 8; ++r) s[t][r] = -1e30f;
      }
    }
    // online softmax per row slot (rows live in C-layout VGPR slots)
#pragma unroll
    for (int r = 0; r < 8; ++r) {
      float v = fmaxf(fmaxf(s[0][r], s[1][r]), fmaxf(s[2][r], s[3][r]));
      v = fmaxf(v, __shfl_xor(v, 1, 32));
      v = fmaxf(v, __shfl_xor(v, 2, 32));
      v = fmaxf(v, __shfl_xor(v, 4, 32));
      v = fmaxf(v, __shfl_xor(v, 8, 32));
      float mn    = fmaxf(m_s[r], v);
      float alpha = __expf(m_s[r] - mn);
      m_s[r] = mn;
      float rs = 0.f;
#pragma unroll
      for (int t = 0; t < 4; ++t) {
        float p = __expf(s[t][r] - mn);
        s[t][r] = p;
        rs += p;
      }
      rs += __shfl_xor(rs, 1, 32);
      rs += __shfl_xor(rs, 2, 32);
      rs += __shfl_xor(rs, 4, 32);
      rs += __shfl_xor(rs, 8, 32);
      l_s[r] = l_s[r] * alpha + rs;
      o0[r] *= alpha;
      o1[r] *= alpha;
    }
    // restage P as f16 into per-wave LDS strip (same-wave LDS ops in-order)
#pragma unroll
    for (int t = 0; t < 4; ++t)
#pragma unroll
      for (int r = 0; r < 8; ++r)
        Pw[(r + 8 * hi) * VST + t * 16 + (lane & 15)] = (_Float16)s[t][r];
    // O += P(16x64) @ V(64x32) : K split into two 32-wide WMMA steps
#pragma unroll
    for (int ks = 0; ks < 2; ++ks) {
      v16h pf = ld_frag(Pw + ks * 32, VST, lane);
      v16h v0 = ld_frag(Vt +            ks * 32, VST, lane);
      v16h v1 = ld_frag(Vt + 16 * VST + ks * 32, VST, lane);
      o0 = wmma16(pf, v0, o0);
      o1 = wmma16(pf, v1, o1);
    }
  }

  // epilogue: normalize and scatter to [q, b, h*32 + d]
#pragma unroll
  for (int r = 0; r < 8; ++r) {
    int q = qb0 + wave * 16 + r + 8 * hi;
    if (q < NQ) {
      float inv = 1.f / l_s[r];
      float* op = Op + ((size_t)q * BSZ + b) * EMB + h * HD;
      op[(lane & 15)]      = o0[r] * inv;
      op[16 + (lane & 15)] = o1[r] * inv;
    }
  }
}

// ---------------- residual add + layernorm (one wave per 256-wide row) ----
__global__ __launch_bounds__(256)
void resid_ln_kernel(const float* __restrict__ a, const float* __restrict__ res,
                     const float* __restrict__ g, const float* __restrict__ beta,
                     float* __restrict__ out, int rows)
{
  const int w    = blockIdx.x * 8 + (threadIdx.x >> 5);
  const int lane = threadIdx.x & 31;
  if (w >= rows) return;
  const float* pa = a   + (size_t)w * EMB;
  const float* pr = res + (size_t)w * EMB;
  float v[8];
  float sum = 0.f;
#pragma unroll
  for (int i = 0; i < 8; ++i) {
    v[i] = pa[lane + 32 * i] + pr[lane + 32 * i];
    sum += v[i];
  }
#pragma unroll
  for (int m = 1; m < 32; m <<= 1) sum += __shfl_xor(sum, m, 32);
  float mu  = sum * (1.f / EMB);
  float var = 0.f;
#pragma unroll
  for (int i = 0; i < 8; ++i) { float d = v[i] - mu; var += d * d; }
#pragma unroll
  for (int m = 1; m < 32; m <<= 1) var += __shfl_xor(var, m, 32);
  float rstd = rsqrtf(var * (1.f / EMB) + LNEPS);
  float* po = out + (size_t)w * EMB;
#pragma unroll
  for (int i = 0; i < 8; ++i) {
    int c = lane + 32 * i;
    po[c] = (v[i] - mu) * rstd * g[c] + beta[c];
  }
}

// ---------------- elementwise add ----------------
__global__ void add_kernel(const float* __restrict__ a, const float* __restrict__ b,
                           float* __restrict__ out, int n)
{
  int i = blockIdx.x * blockDim.x + threadIdx.x;
  if (i < n) out[i] = a[i] + b[i];
}

// ---------------- deformable sampling: softmax(aw) + bilinear gather ------
// one wave per (q, b, head); lane = head-dim channel -> coalesced f16 taps.
__global__ __launch_bounds__(256)
void deform_sample_kernel(const _Float16* __restrict__ valp,
                          const float* __restrict__ off,
                          const float* __restrict__ awl,
                          const float* __restrict__ refp,
                          float* __restrict__ out,
                          const int* __restrict__ pH, const int* __restrict__ pW)
{
  const int H    = *pH;
  const int W    = *pW;
  const int gw   = (blockIdx.x * blockDim.x + threadIdx.x) >> 5;
  const int lane = threadIdx.x & 31;
  const int h    = gw & 7;
  const int qb   = gw >> 3;          // row = q*BSZ + b
  const int b    = qb & (BSZ - 1);
  const int q    = qb >> 4;

  // per-(q,b,h) softmax over 4 points (uniform across lanes)
  float a[NP];
  float mx = -1e30f;
#pragma unroll
  for (int p = 0; p < NP; ++p) {
    a[p] = awl[(size_t)qb * (NHD * NP) + h * NP + p];
    mx = fmaxf(mx, a[p]);
  }
  float ssum = 0.f;
#pragma unroll
  for (int p = 0; p < NP; ++p) { a[p] = __expf(a[p] - mx); ssum += a[p]; }
  const float inv = 1.f / ssum;

  const float rx = refp[((size_t)b * NQ + q) * 2 + 0];
  const float ry = refp[((size_t)b * NQ + q) * 2 + 1];
  const float* offp = off + (size_t)qb * (NHD * NP * 2) + h * NP * 2;

  float acc = 0.f;
#pragma unroll
  for (int p = 0; p < NP; ++p) {
    float lx = rx + offp[p * 2 + 0] / (float)W;
    float ly = ry + offp[p * 2 + 1] / (float)H;
    // grid = 2*loc - 1 ; x = (gx+1)*W/2 - 0.5 == loc_x*W - 0.5
    float x = lx * (float)W - 0.5f;
    float y = ly * (float)H - 0.5f;
    float x0f = floorf(x), y0f = floorf(y);
    int   x0 = (int)x0f,   y0 = (int)y0f;
    float wx = x - x0f,    wy = y - y0f;
    float bil = 0.f;
#pragma unroll
    for (int dy = 0; dy < 2; ++dy)
#pragma unroll
      for (int dx = 0; dx < 2; ++dx) {
        int xi = x0 + dx, yi = y0 + dy;
        bool ok = (xi >= 0) & (xi < W) & (yi >= 0) & (yi < H);
        int xc = min(max(xi, 0), W - 1);
        int yc = min(max(yi, 0), H - 1);
        float wgt = (dx ? wx : 1.f - wx) * (dy ? wy : 1.f - wy);
        size_t idx = (((size_t)yc * W + xc) * BSZ + b) * EMB + h * HD + lane;
        float v = (float)valp[idx];           // unconditional, batchable
        bil += (ok ? wgt : 0.f) * v;
      }
    acc += a[p] * inv * bil;
  }
  out[(size_t)qb * EMB + h * HD + lane] = acc;
}

// ---------------- host orchestration ----------------
static inline void launch_gemm(const float* A, const float* W, const float* b,
                               void* C, int M, int N, int K, int relu,
                               int half_out, hipStream_t s)
{
  dim3 grid((M + BM - 1) / BM, (N + BN - 1) / BN);
  gemm_f16_kernel<<<grid, 256, 0, s>>>(A, W, b, C, M, N, K, relu, half_out);
}

extern "C" void kernel_launch(void* const* d_in, const int* in_sizes, int n_in,
                              void* d_out, int out_size, void* d_ws, size_t ws_size,
                              hipStream_t stream)
{
  (void)in_sizes; (void)n_in; (void)out_size; (void)ws_size;

  const float* query     = (const float*)d_in[0];
  const float* query_pos = (const float*)d_in[1];
  const float* value     = (const float*)d_in[2];
  const float* refp      = (const float*)d_in[3];
  const float* in_proj_w = (const float*)d_in[4];
  const float* in_proj_b = (const float*)d_in[5];
  const float* out_proj_w= (const float*)d_in[6];
  const float* out_proj_b= (const float*)d_in[7];
  const float* so_w      = (const float*)d_in[8];
  const float* so_b      = (const float*)d_in[9];
  const float* aw_w      = (const float*)d_in[10];
  const float* aw_b      = (const float*)d_in[11];
  const float* vp_w      = (const float*)d_in[12];
  const float* vp_b      = (const float*)d_in[13];
  const float* op_w      = (const float*)d_in[14];
  const float* op_b      = (const float*)d_in[15];
  const float* ffn_w1    = (const float*)d_in[16];
  const float* ffn_b1    = (const float*)d_in[17];
  const float* ffn_w2    = (const float*)d_in[18];
  const float* ffn_b2    = (const float*)d_in[19];
  const float* ln0_g     = (const float*)d_in[20];
  const float* ln0_b     = (const float*)d_in[21];
  const float* ln1_g     = (const float*)d_in[22];
  const float* ln1_b     = (const float*)d_in[23];
  const float* ln2_g     = (const float*)d_in[24];
  const float* ln2_b     = (const float*)d_in[25];
  const int*   pH        = (const int*)d_in[26];
  const int*   pW        = (const int*)d_in[27];

  float* ws = (float*)d_ws;
  const size_t SZ = (size_t)NQB * EMB;            // 3,686,400 floats
  float*    P1 = ws;
  float*    P2 = ws + SZ;
  float*    P3 = ws + 2 * SZ;
  float*    P4 = ws + 3 * SZ;
  _Float16* PV = (_Float16*)(ws + 4 * SZ);        // 160000*256 f16

  const int nAdd = (int)SZ;
  const int addGrid = (nAdd + 255) / 256;

  // ---- self attention ----
  add_kernel<<<addGrid, 256, 0, stream>>>(query, query_pos, P1, nAdd);
  launch_gemm(P1,    in_proj_w,                     in_proj_b,           P2, NQB, EMB, EMB, 0, 0, stream);
  launch_gemm(P1,    in_proj_w + (size_t)EMB*EMB,   in_proj_b + EMB,     P3, NQB, EMB, EMB, 0, 0, stream);
  launch_gemm(query, in_proj_w + (size_t)2*EMB*EMB, in_proj_b + 2*EMB,   P4, NQB, EMB, EMB, 0, 0, stream);
  flash_attn_kernel<<<dim3(BSZ * NHD, (NQ + QBLK - 1) / QBLK), 128, 0, stream>>>(P2, P3, P4, P1);
  launch_gemm(P1, out_proj_w, out_proj_b, P2, NQB, EMB, EMB, 0, 0, stream);
  resid_ln_kernel<<<NQB / 8, 256, 0, stream>>>(P2, query, ln0_g, ln0_b, P1, NQB);

  // ---- deformable cross attention ----
  add_kernel<<<addGrid, 256, 0, stream>>>(P1, query_pos, P2, nAdd);
  launch_gemm(value, vp_w, vp_b, PV, NVAL * BSZ, EMB, EMB, 0, 1, stream);
  float* OFF = P3;
  float* AWL = P3 + (size_t)NQB * 64;
  launch_gemm(P2, so_w, so_b, OFF, NQB, NHD * NP * 2, EMB, 0, 0, stream);
  launch_gemm(P2, aw_w, aw_b, AWL, NQB, NHD * NP,     EMB, 0, 0, stream);
  deform_sample_kernel<<<NQB, 256, 0, stream>>>(PV, OFF, AWL, refp, P4, pH, pW);
  launch_gemm(P4, op_w, op_b, P2, NQB, EMB, EMB, 0, 0, stream);
  resid_ln_kernel<<<NQB / 8, 256, 0, stream>>>(P2, P1, ln1_g, ln1_b, P4, NQB);

  // ---- FFN ----
  launch_gemm(P4, ffn_w1, ffn_b1, P1, NQB, DFF, EMB, 1, 0, stream);   // spans P1..P2
  launch_gemm(P1, ffn_w2, ffn_b2, P3, NQB, EMB, DFF, 0, 0, stream);
  resid_ln_kernel<<<NQB / 8, 256, 0, stream>>>(P3, P4, ln2_g, ln2_b, (float*)d_out, NQB);
}
